// SpatialSelfAttentionModel_61203283968630
// MI455X (gfx1250) — matrix operand order repaired and checked by reading, not compiled, not run on hardware
//
#include <hip/hip_runtime.h>
#include <hip/hip_bf16.h>

typedef __bf16 bf16;
typedef __attribute__((ext_vector_type(16))) __bf16 bf16x16;
typedef __attribute__((ext_vector_type(8)))  float   f32x8;
typedef __attribute__((ext_vector_type(2)))  unsigned int u32x2;

#define LDS_STRIDE 48   // 32 k-values + 16 pad (bf16); row = 96 B, 16B-aligned

union FragAB { u32x2 q[4]; bf16x16 v; };

// 32-byte contiguous global -> LDS async copy (2x b128), ASYNCcnt-tracked.
__device__ __forceinline__ void async_cp32(bf16* lds_dst, const bf16* gsrc)
{
    unsigned l = (unsigned)(uintptr_t)lds_dst;   // low 32 bits == LDS offset
    asm volatile("global_load_async_to_lds_b128 %0, %1, off"
                 :: "v"(l), "v"(gsrc) : "memory");
    asm volatile("global_load_async_to_lds_b128 %0, %1, off offset:16"
                 :: "v"(l), "v"(gsrc) : "memory");
}

// ---------------------------------------------------------------------------
// bf16 WMMA GEMM:  C[b][M][N] = A[b][M][K] * B[b][N][K]^T  (+ bias)
//   BIAS_MODE: 0 = none, 1 = bias[n] (per output column), 2 = bias[m] (row)
// Block tile 128x128, 8 wave32 waves, each 32x64 (2x4 16x16x32 frags).
// Double-buffered LDS fed by global_load_async_to_lds_b128.
// M,N multiples of 128; K multiple of 32.
// ---------------------------------------------------------------------------
template<int BIAS_MODE, bool OUT_BF16>
__global__ __launch_bounds__(256)
void gemm_bf16(const bf16* __restrict__ Ag, const bf16* __restrict__ Bg,
               const float* __restrict__ bias, void* __restrict__ Cg,
               int M, int N, int K,
               long long sA, long long sB, long long sC)
{
    __shared__ bf16 As[2][128][LDS_STRIDE];   // As[buf][m][k]
    __shared__ bf16 Bs[2][128][LDS_STRIDE];   // Bs[buf][n][k]

    const int b      = blockIdx.y;
    const int tilesN = N >> 7;
    const int m0     = (blockIdx.x / tilesN) << 7;
    const int n0     = (blockIdx.x % tilesN) << 7;

    const bf16* A = Ag + (long long)b * sA;
    const bf16* B = Bg + (long long)b * sB;

    const int t    = threadIdx.x;
    const int lane = t & 31;
    const int wave = t >> 5;
    const int lh   = lane & 15;   // 0..15
    const int lg   = lane >> 4;   // 0/1
    const int wm   = wave & 3;    // 4 wave-rows of 32
    const int wn   = wave >> 2;   // 2 wave-cols of 64

    // per-thread copy slice: one row, 32 bytes (k-contiguous)
    const int rowL = t >> 1;
    const int colL = (t & 1) * 16;
    const bf16* Asrc = A + (size_t)(m0 + rowL) * K + colL;
    const bf16* Bsrc = B + (size_t)(n0 + rowL) * K + colL;

    f32x8 acc[2][4];
    #pragma unroll
    for (int i = 0; i < 2; ++i)
        #pragma unroll
        for (int j = 0; j < 4; ++j)
            #pragma unroll
            for (int e = 0; e < 8; ++e) acc[i][j][e] = 0.0f;

    // prefetch first K-tile into buffer 0
    async_cp32(&As[0][rowL][colL], Asrc);
    async_cp32(&Bs[0][rowL][colL], Bsrc);

    for (int kt = 0; kt < K; kt += 32) {
        const int buf = (kt >> 5) & 1;
        if (kt + 32 < K) {
            // prefetch next tile into the other buffer (prev compute barrier
            // guarantees nobody is still reading it)
            async_cp32(&As[buf ^ 1][rowL][colL], Asrc + kt + 32);
            async_cp32(&Bs[buf ^ 1][rowL][colL], Bsrc + kt + 32);
            asm volatile("s_wait_asynccnt 0x4" ::: "memory"); // current tile done
        } else {
            asm volatile("s_wait_asynccnt 0x0" ::: "memory");
        }
        __syncthreads();

        FragAB aF[2], bF[4];
        #pragma unroll
        for (int fi = 0; fi < 2; ++fi) {
            // lane<16: M=lh, K {0..7,16..23}; lane>=16: K {8..15,24..31}
            const bf16* ar = &As[buf][wm * 32 + fi * 16 + lh][lg * 8];
            aF[fi].q[0] = ((const u32x2*)(ar     ))[0];
            aF[fi].q[1] = ((const u32x2*)(ar + 4 ))[0];
            aF[fi].q[2] = ((const u32x2*)(ar + 16))[0];
            aF[fi].q[3] = ((const u32x2*)(ar + 20))[0];
        }
        #pragma unroll
        for (int fj = 0; fj < 4; ++fj) {
            // col = lh; lane group lg covers K = lg*8 + {0..7, 16..23}
            const bf16* br = &Bs[buf][wn * 64 + fj * 16 + lh][lg * 8];
            bF[fj].q[0] = ((const u32x2*)(br     ))[0];
            bF[fj].q[1] = ((const u32x2*)(br + 4 ))[0];
            bF[fj].q[2] = ((const u32x2*)(br + 16))[0];
            bF[fj].q[3] = ((const u32x2*)(br + 20))[0];
        }
        #pragma unroll
        for (int fi = 0; fi < 2; ++fi)
            #pragma unroll
            for (int fj = 0; fj < 4; ++fj)
                acc[fi][fj] = __builtin_amdgcn_wmma_f32_16x16x32_bf16(
                    false, aF[fi].v, false, bF[fj].v,
                    (short)0, acc[fi][fj], false, false);

        __syncthreads();
    }

    // epilogue: lane lh = n, vgpr r + lg*8 = m within each 16x16 frag
    const size_t cb = (size_t)b * (size_t)sC;
    #pragma unroll
    for (int fi = 0; fi < 2; ++fi) {
        #pragma unroll
        for (int fj = 0; fj < 4; ++fj) {
            const int nIdx = n0 + wn * 64 + fj * 16 + lh;
            const float cbias = (BIAS_MODE == 1) ? bias[nIdx] : 0.0f;
            #pragma unroll
            for (int r = 0; r < 8; ++r) {
                const int mIdx = m0 + wm * 32 + fi * 16 + lg * 8 + r;
                float v = acc[fi][fj][r] + cbias;
                if (BIAS_MODE == 2) v += bias[mIdx];
                const size_t o = cb + (size_t)mIdx * N + nIdx;
                if (OUT_BF16) ((bf16*)Cg)[o] = (bf16)v;
                else          ((float*)Cg)[o] = v;
            }
        }
    }
}

// ---------------------------------------------------------------------------
// Row softmax over last axis, f32 in -> bf16 out.  grid = (1024, B)
// ---------------------------------------------------------------------------
__global__ __launch_bounds__(256)
void softmax_rows(const float* __restrict__ S, bf16* __restrict__ Aout, int N)
{
    __shared__ float red[256];
    const size_t rbase = ((size_t)blockIdx.y * gridDim.x + blockIdx.x) * N;
    const float* row = S + rbase;
    bf16* orow = Aout + rbase;
    const int t = threadIdx.x;

    float v[4];
    float mx = -3.4e38f;
    #pragma unroll
    for (int i = 0; i < 4; ++i) { v[i] = row[t + i * 256]; mx = fmaxf(mx, v[i]); }
    red[t] = mx; __syncthreads();
    for (int s = 128; s > 0; s >>= 1) { if (t < s) red[t] = fmaxf(red[t], red[t + s]); __syncthreads(); }
    mx = red[0]; __syncthreads();

    float sum = 0.f;
    #pragma unroll
    for (int i = 0; i < 4; ++i) { v[i] = __expf(v[i] - mx); sum += v[i]; }
    red[t] = sum; __syncthreads();
    for (int s = 128; s > 0; s >>= 1) { if (t < s) red[t] += red[t + s]; __syncthreads(); }
    const float inv = 1.0f / red[0];
    #pragma unroll
    for (int i = 0; i < 4; ++i) orow[t + i * 256] = (bf16)(v[i] * inv);
}

// ---------------------------------------------------------------------------
// patchify (token-major): Xp[b][n=i*32+j][k=c*256+u*16+v] = x[b][c][i*16+u][j*16+v]
// One block per x row (b,c,i,u): coalesced read, 32B-segment writes.
// ---------------------------------------------------------------------------
__global__ __launch_bounds__(256)
void patchify(const float* __restrict__ x, bf16* __restrict__ Xp)
{
    int rid = blockIdx.x;
    const int u = rid & 15;  rid >>= 4;
    const int i = rid & 31;  rid >>= 5;
    const int c = rid % 3;
    const int b = rid / 3;
    const float* src = x + (((size_t)(b * 3 + c) * 512) + i * 16 + u) * 512;
    const int t = threadIdx.x;
    #pragma unroll
    for (int e0 = 0; e0 < 2; ++e0) {
        const int e = t + e0 * 256;
        const int j = e >> 4, v = e & 15;
        Xp[(((size_t)b * 1024) + i * 32 + j) * 768 + c * 256 + u * 16 + v] =
            (bf16)src[e];
    }
}

// ---------------------------------------------------------------------------
// unpatchify (token-major in): y[b][c][i*16+u][j*16+v] = Yf[b][n][k] + b_tc[c]
// ---------------------------------------------------------------------------
__global__ __launch_bounds__(256)
void unpatchify(const float* __restrict__ Yf, const float* __restrict__ b_tc,
                float* __restrict__ y)
{
    int rid = blockIdx.x;
    const int u = rid & 15;  rid >>= 4;
    const int i = rid & 31;  rid >>= 5;
    const int c = rid % 3;
    const int b = rid / 3;
    const int t = threadIdx.x;
    const float bias = b_tc[c];
    float* dst = y + (((size_t)(b * 3 + c) * 512) + i * 16 + u) * 512;
    #pragma unroll
    for (int e0 = 0; e0 < 2; ++e0) {
        const int e = t + e0 * 256;
        const int j = e >> 4, v = e & 15;
        dst[e] = Yf[(((size_t)b * 1024) + i * 32 + j) * 768 + c * 256 + u * 16 + v]
                 + bias;
    }
}

// ---------------------------------------------------------------------------
__global__ __launch_bounds__(256)
void cvt_f32_bf16(const float* __restrict__ s, bf16* __restrict__ d, int n)
{
    const int i = blockIdx.x * 256 + threadIdx.x;
    if (i < n) d[i] = (bf16)s[i];
}

__global__ __launch_bounds__(256)
void transpose_cvt(const float* __restrict__ s, bf16* __restrict__ d, int R, int C_)
{
    const int i = blockIdx.x * 256 + threadIdx.x;
    if (i < R * C_) {
        const int r = i / C_, c = i % C_;
        d[(size_t)c * R + r] = (bf16)s[i];   // d[C][R] = s[R][C]^T
    }
}

// ---------------------------------------------------------------------------
extern "C" void kernel_launch(void* const* d_in, const int* in_sizes, int n_in,
                              void* d_out, int out_size, void* d_ws, size_t ws_size,
                              hipStream_t stream)
{
    (void)in_sizes; (void)n_in; (void)out_size; (void)ws_size;
    const float* x       = (const float*)d_in[0];
    const float* W_pe    = (const float*)d_in[1];
    const float* b_pe    = (const float*)d_in[2];
    const float* W_theta = (const float*)d_in[3];
    const float* b_theta = (const float*)d_in[4];
    const float* W_phi   = (const float*)d_in[5];
    const float* b_phi   = (const float*)d_in[6];
    const float* W_g     = (const float*)d_in[7];
    const float* b_g     = (const float*)d_in[8];
    const float* W_o     = (const float*)d_in[9];
    const float* b_o     = (const float*)d_in[10];
    const float* W_tc    = (const float*)d_in[11];
    const float* b_tc    = (const float*)d_in[12];

    char* ws = (char*)d_ws;
    size_t off = 0;
    auto alloc = [&](size_t bytes) -> void* {
        void* p = ws + off;
        off += (bytes + 255) & ~(size_t)255;
        return p;
    };

    const int    E  = 768;
    const size_t W2 = (size_t)E * E;        // 589824 (weights, also [N][K] for GEMM)
    const size_t TN = (size_t)1024 * E;     // token-major activation per batch
    const size_t SN = (size_t)1024 * 1024;  // scores per batch

    bf16* Wpe_bf  = (bf16*)alloc(W2 * 2);
    bf16* Wth_bf  = (bf16*)alloc(W2 * 2);
    bf16* Wph_bf  = (bf16*)alloc(W2 * 2);
    bf16* Wg_bf   = (bf16*)alloc(W2 * 2);
    bf16* Wo_bf   = (bf16*)alloc(W2 * 2);
    bf16* Wtc_t   = (bf16*)alloc(W2 * 2);      // W_tc^T : [k_out][e]
    bf16* Xp      = (bf16*)alloc(32 * TN * 2); // patches   [b][n][k]
    bf16* t_bf    = (bf16*)alloc(32 * TN * 2); // embeddings[b][n][e]
    bf16* theta   = (bf16*)alloc(32 * TN * 2); // [b][n][e]
    bf16* phi     = (bf16*)alloc(32 * TN * 2); // [b][n][e]
    bf16* g_ch    = (bf16*)alloc(32 * TN * 2); // channel-major [b][e][m]
    float* scores = (float*)alloc(32 * SN * 4);
    bf16* attn    = (bf16*)alloc(32 * SN * 2);
    // reuse dead regions:
    bf16*  out_bf  = Xp;       // Xp dead after patch-embed GEMM
    bf16*  out2_bf = t_bf;     // t dead after theta/phi/g GEMMs
    float* yflat   = scores;   // scores dead after softmax

    const int wBlocks = (int)((W2 + 255) / 256);
    cvt_f32_bf16<<<wBlocks, 256, 0, stream>>>(W_pe,    Wpe_bf, (int)W2);
    cvt_f32_bf16<<<wBlocks, 256, 0, stream>>>(W_theta, Wth_bf, (int)W2);
    cvt_f32_bf16<<<wBlocks, 256, 0, stream>>>(W_phi,   Wph_bf, (int)W2);
    cvt_f32_bf16<<<wBlocks, 256, 0, stream>>>(W_g,     Wg_bf,  (int)W2);
    cvt_f32_bf16<<<wBlocks, 256, 0, stream>>>(W_o,     Wo_bf,  (int)W2);
    transpose_cvt<<<wBlocks, 256, 0, stream>>>(W_tc,   Wtc_t, E, E);

    patchify<<<32 * 3 * 32 * 16, 256, 0, stream>>>(x, Xp);

    const dim3 gA((1024 / 128) * (768 / 128), 32);    // M=1024, N=768
    const dim3 gG((768 / 128) * (1024 / 128), 32);    // M=768,  N=1024
    const dim3 gS((1024 / 128) * (1024 / 128), 32);   // M=1024, N=1024

    // t[n][e] = Xp[n][k] . W_pe[e][k]^T + b_pe[e]
    gemm_bf16<1, true><<<gA, 256, 0, stream>>>(
        Xp, Wpe_bf, b_pe, t_bf, 1024, 768, 768, (long long)TN, 0, (long long)TN);
    // theta/phi[n][e] = t[n][k] . W[e][k]^T + b[e]
    gemm_bf16<1, true><<<gA, 256, 0, stream>>>(
        t_bf, Wth_bf, b_theta, theta, 1024, 768, 768, (long long)TN, 0, (long long)TN);
    gemm_bf16<1, true><<<gA, 256, 0, stream>>>(
        t_bf, Wph_bf, b_phi, phi, 1024, 768, 768, (long long)TN, 0, (long long)TN);
    // g_ch[e][m] = W_g[e][k] . t[m][k]^T + b_g[e]   (row bias)
    gemm_bf16<2, true><<<gG, 256, 0, stream>>>(
        Wg_bf, t_bf, b_g, g_ch, 768, 1024, 768, 0, (long long)TN, (long long)TN);
    // scores[n][m] = theta[n][e] . phi[m][e]^T
    gemm_bf16<0, false><<<gS, 256, 0, stream>>>(
        theta, phi, nullptr, scores, 1024, 1024, 768,
        (long long)TN, (long long)TN, (long long)SN);
    // attn = softmax_m(scores)
    softmax_rows<<<dim3(1024, 32), 256, 0, stream>>>(scores, attn, 1024);
    // out[n][e] = attn[n][m] . g_ch[e][m]^T
    gemm_bf16<0, true><<<gA, 256, 0, stream>>>(
        attn, g_ch, nullptr, out_bf, 1024, 768, 1024,
        (long long)SN, (long long)TN, (long long)TN);
    // out2[n][eo] = out[n][e] . W_o[eo][e]^T + b_o[eo]
    gemm_bf16<1, true><<<gA, 256, 0, stream>>>(
        out_bf, Wo_bf, b_o, out2_bf, 1024, 768, 768, (long long)TN, 0, (long long)TN);
    // yflat[n][k] = out2[n][e] . Wtc_t[k][e]^T
    gemm_bf16<0, false><<<gA, 256, 0, stream>>>(
        out2_bf, Wtc_t, nullptr, yflat, 1024, 768, 768, (long long)TN, 0, (long long)TN);
    // y = scatter(yflat) + b_tc
    unpatchify<<<32 * 3 * 32 * 16, 256, 0, stream>>>(yflat, b_tc, (float*)d_out);
}
